// TaskSpecificMLPs_10514079941541
// MI455X (gfx1250) — matrix operand order repaired and checked by reading, not compiled, not run on hardware
//
#include <hip/hip_runtime.h>
#include <hip/hip_bf16.h>

typedef __bf16 v16bf __attribute__((ext_vector_type(16)));
typedef float  v8f   __attribute__((ext_vector_type(8)));
typedef __bf16 bf16x4 __attribute__((ext_vector_type(4)));
typedef unsigned int u32x4 __attribute__((ext_vector_type(4)));
typedef int i32x4 __attribute__((ext_vector_type(4)));
typedef int i32x8 __attribute__((ext_vector_type(8)));

constexpr int kN = 4096;   // samples
constexpr int kD = 1024;   // in features
constexpr int kB = 512;    // bottleneck
constexpr int kE = 16;     // experts

// ---------------------------------------------------------------- TDM shim
// therock-10.0 headers ship the TDM descriptor header and a 6-arg builtin;
// ROCm 7.2 has the 5-arg form. Arity-switch on header presence.
#if __has_include(<hip/amd_detail/amd_gfx1250_TDM.h>)
#define TDM_SIX_ARGS 1
#endif

__device__ inline void tdm_load_to_lds(u32x4 g0, i32x8 g1, i32x4 g2, i32x4 g3) {
#ifdef TDM_SIX_ARGS
  i32x8 g4 = {};
  __builtin_amdgcn_tensor_load_to_lds(g0, g1, g2, g3, g4, 0);
#else
  __builtin_amdgcn_tensor_load_to_lds(g0, g1, g2, g3, 0);
#endif
}

union Frag {
  v16bf v;
  uint4 q[2];
};

__device__ inline v8f wmma_bf16(const v16bf& a, const v16bf& b, v8f c) {
  return __builtin_amdgcn_wmma_f32_16x16x32_bf16(false, a, false, b,
                                                 (short)0, c, false, false);
}

// Fused bias + relu + layer-2 partial dot; width-16 butterfly leaves the
// per-row sum in all lanes of each half-wave group.
__device__ inline void epilogue_tile(const v8f& acc, float b1v, float w2v,
                                     float* rs) {
  #pragma unroll
  for (int r = 0; r < 8; ++r) {
    float h = fmaxf(acc[r] + b1v, 0.0f);
    float p = h * w2v;
    #pragma unroll
    for (int off = 8; off > 0; off >>= 1) p += __shfl_xor(p, off, 16);
    rs[r] += p;
  }
}

// ---------------------------------------------------------------- utility
__global__ void init_zero_kernel(float* __restrict__ out, int* __restrict__ counts) {
  int i = blockIdx.x * blockDim.x + threadIdx.x;
  if (i < kN + 1) out[i] = 0.0f;
  if (i < kE) counts[i] = 0;
}

__global__ void scatter_kernel(const int* __restrict__ keys, int* __restrict__ counts,
                               int* __restrict__ bucket) {
  int i = blockIdx.x * blockDim.x + threadIdx.x;
  if (i < kN) {
    int e = keys[i];
    int pos = atomicAdd(&counts[e], 1);
    bucket[e * kN + pos] = i;
  }
}

// fp32 -> bf16 hi/lo split, 4 elements per thread (float4 in, 2x bf16x4 out).
__global__ void convert_kernel(const float* __restrict__ src,
                               __bf16* __restrict__ hi, __bf16* __restrict__ lo,
                               int n4) {
  int i = blockIdx.x * blockDim.x + threadIdx.x;
  if (i < n4) {
    float4 v = ((const float4*)src)[i];
    bf16x4 h, l;
    h[0] = (__bf16)v.x; l[0] = (__bf16)(v.x - (float)h[0]);
    h[1] = (__bf16)v.y; l[1] = (__bf16)(v.y - (float)h[1]);
    h[2] = (__bf16)v.z; l[2] = (__bf16)(v.z - (float)h[2]);
    h[3] = (__bf16)v.w; l[3] = (__bf16)(v.w - (float)h[3]);
    ((bf16x4*)hi)[i] = h;
    ((bf16x4*)lo)[i] = l;
  }
}

// Per-expert L1 of params, 64 fixed-order partials per expert.
__global__ void l1_reduce_kernel(const float* __restrict__ W1, const float* __restrict__ b1,
                                 const float* __restrict__ W2, const float* __restrict__ b2,
                                 float* __restrict__ l1part) {
  const int ex = blockIdx.y, chunk = blockIdx.x, t = threadIdx.x;
  const float* base = W1 + (size_t)ex * (kB * kD) + chunk * 8192;
  float s = 0.0f;
  #pragma unroll
  for (int i = 0; i < 8; ++i) {
    float4 v = *(const float4*)(base + i * 1024 + t * 4);
    s += fabsf(v.x) + fabsf(v.y) + fabsf(v.z) + fabsf(v.w);
  }
  if (chunk == 0) {
    s += fabsf(b1[ex * kB + t]) + fabsf(b1[ex * kB + 256 + t]);
    s += fabsf(W2[ex * kB + t]) + fabsf(W2[ex * kB + 256 + t]);
    if (t == 0) s += fabsf(b2[ex]);
  }
  __shared__ float red[256];
  red[t] = s;
  __syncthreads();
  for (int off = 128; off > 0; off >>= 1) {
    if (t < off) red[t] += red[t + off];
    __syncthreads();
  }
  if (t == 0) l1part[ex * 64 + chunk] = red[0];
}

// ------------------------------------------------------- fast path (preconv)
// One wave per 16-sample tile. TDM gather-mode pulls the 16 (scattered)
// sample rows of x_hi/x_lo into LDS in one shot each; inner loop is pure
// ds_load_b128 / global b128 loads + 12 WMMAs per K-step (4 n-tiles blocked).
__global__ void __launch_bounds__(32)
moe_gemm_pre_kernel(const __bf16* __restrict__ xh, const __bf16* __restrict__ xl,
                    const __bf16* __restrict__ w1h, const __bf16* __restrict__ w1l,
                    const float* __restrict__ b1, const float* __restrict__ W2,
                    const float* __restrict__ b2, const int* __restrict__ counts,
                    const int* __restrict__ bucket, float* __restrict__ out) {
  extern __shared__ __bf16 smem[];          // 2 * 16*1024 bf16 = 64 KB
  __bf16* As_hi = smem;
  __bf16* As_lo = smem + 16 * kD;

  const int mtile = blockIdx.x;
  const int ex    = blockIdx.y;
  const int cnt   = counts[ex];
  if (mtile * 16 >= cnt) return;            // uniform early exit

  const int lane = threadIdx.x;
  const int rowA = lane & 15;
  const int half = lane >> 4;

  // Wave-uniform row indices for the TDM gather descriptor.
  int idx[16];
  #pragma unroll
  for (int j = 0; j < 16; ++j)
    idx[j] = bucket[ex * kN + min(mtile * 16 + j, cnt - 1)];

  // D# group1 (shared by both gathers): data_size=2B, tensor = [kN, kD],
  // tile_dim0 = kD (full row), tile_dim1 = 16 valid indices, stride0 = kD.
  i32x8 g1;
  g1[0] = 0x00010000;                                   // data_size=1 (2B)
  g1[1] = (int)((kD & 0xFFFF) << 16);                   // tensor_dim0 lo16
  g1[2] = (int)((kD >> 16) | ((kN & 0xFFFF) << 16));    // dim0 hi | dim1 lo
  g1[3] = (int)((kN >> 16) | (kD << 16));               // dim1 hi | tile_dim0
  g1[4] = 16;                                           // tile_dim1 = 16 idx
  g1[5] = kD;                                           // tensor_dim0_stride
  g1[6] = 0;
  g1[7] = 0;
  // Row indices: 16-bit packed, 8 per group.
  i32x4 g2, g3;
  #pragma unroll
  for (int j = 0; j < 4; ++j) {
    g2[j] = (idx[2 * j] & 0xFFFF) | (idx[2 * j + 1] << 16);
    g3[j] = (idx[8 + 2 * j] & 0xFFFF) | (idx[8 + 2 * j + 1] << 16);
  }
  // D# group0 per tensor: count=1, gather_mode, 16-bit indices, type=2.
  unsigned long long gah = (unsigned long long)(uintptr_t)xh;
  unsigned long long gal = (unsigned long long)(uintptr_t)xl;
  u32x4 g0h, g0l;
  g0h[0] = 0x80000001u;
  g0h[1] = (unsigned)(uintptr_t)As_hi;                  // lds_addr (low 32b)
  g0h[2] = (unsigned)(gah & 0xFFFFFFFFu);
  g0h[3] = (unsigned)((gah >> 32) & 0x1FFFFFFu) | 0x80000000u;
  g0l[0] = 0x80000001u;
  g0l[1] = (unsigned)(uintptr_t)As_lo;
  g0l[2] = (unsigned)(gal & 0xFFFFFFFFu);
  g0l[3] = (unsigned)((gal >> 32) & 0x1FFFFFFu) | 0x80000000u;

  tdm_load_to_lds(g0h, g1, g2, g3);
  tdm_load_to_lds(g0l, g1, g2, g3);

  const int aK  = half ? 8 : 0;    // A frag K base per half-wave (ISA 7.12.2)
  const int bKo = half ? 8 : 0;    // B frag K base per half-wave

  float rs[8];
  #pragma unroll
  for (int r = 0; r < 8; ++r) rs[r] = 0.0f;

  __builtin_amdgcn_s_wait_tensorcnt(0);
  asm volatile("" ::: "memory");   // keep ds reads below the tensorcnt wait

  for (int ntb = 0; ntb < kB / 16; ntb += 4) {
    v8f acc[4] = {};
    const __bf16 *wph[4], *wpl[4];
    #pragma unroll
    for (int j = 0; j < 4; ++j) {
      int nn = (ntb + j) * 16 + rowA;
      size_t off = ((size_t)(ex * kB + nn)) * kD + bKo;
      wph[j] = w1h + off;
      wpl[j] = w1l + off;
    }
    for (int k0 = 0; k0 < kD; k0 += 32) {
      Frag ah, al;
      const __bf16* pah = As_hi + rowA * kD + k0 + aK;
      const __bf16* pal = As_lo + rowA * kD + k0 + aK;
      ah.q[0] = *(const uint4*)(pah);
      ah.q[1] = *(const uint4*)(pah + 16);
      al.q[0] = *(const uint4*)(pal);
      al.q[1] = *(const uint4*)(pal + 16);
      #pragma unroll
      for (int j = 0; j < 4; ++j) {
        Frag bh, bl;
        const __bf16* pbh = wph[j] + k0;
        const __bf16* pbl = wpl[j] + k0;
        bh.q[0] = *(const uint4*)(pbh);
        bh.q[1] = *(const uint4*)(pbh + 16);
        bl.q[0] = *(const uint4*)(pbl);
        bl.q[1] = *(const uint4*)(pbl + 16);
        acc[j] = wmma_bf16(ah.v, bh.v, acc[j]);
        acc[j] = wmma_bf16(ah.v, bl.v, acc[j]);
        acc[j] = wmma_bf16(al.v, bh.v, acc[j]);
      }
    }
    #pragma unroll
    for (int j = 0; j < 4; ++j) {
      int nn = (ntb + j) * 16 + rowA;
      epilogue_tile(acc[j], b1[ex * kB + nn], W2[ex * kB + nn], rs);
    }
  }

  if ((lane & 15) == 0) {
    const float b2v = b2[ex];
    #pragma unroll
    for (int r = 0; r < 8; ++r) {
      int m = mtile * 16 + half * 8 + r;
      if (m < cnt) out[bucket[ex * kN + m]] = rs[r] + b2v;
    }
  }
}

// ---------------------------------------------- fallback path (convert on fly)
__global__ void __launch_bounds__(32)
moe_gemm_fly_kernel(const float* __restrict__ x, const float* __restrict__ W1,
                    const float* __restrict__ b1, const float* __restrict__ W2,
                    const float* __restrict__ b2, const int* __restrict__ counts,
                    const int* __restrict__ bucket, float* __restrict__ out) {
  const int mtile = blockIdx.x;
  const int ex    = blockIdx.y;
  const int cnt   = counts[ex];
  if (mtile * 16 >= cnt) return;

  const int lane = threadIdx.x;
  const int rowA = lane & 15;
  const int half = lane >> 4;

  const int m_a      = min(mtile * 16 + rowA, cnt - 1);
  const int sample_a = bucket[ex * kN + m_a];
  const float* xrow  = x + (size_t)sample_a * kD;

  const int aK = half ? 8 : 0;
  const int bK = half ? 8 : 0;

  float rs[8];
  #pragma unroll
  for (int r = 0; r < 8; ++r) rs[r] = 0.0f;

  for (int nt = 0; nt < kB / 16; ++nt) {
    const int nn = nt * 16 + rowA;
    const float* wrow = W1 + ((size_t)(ex * kB + nn)) * kD + bK;
    v8f acc = {};
    for (int k0 = 0; k0 < kD; k0 += 32) {
      const float* ap = xrow + k0 + aK;
      float4 a0 = *(const float4*)(ap);
      float4 a1 = *(const float4*)(ap + 4);
      float4 a2 = *(const float4*)(ap + 16);
      float4 a3 = *(const float4*)(ap + 20);
      const float* bp = wrow + k0;
      __builtin_prefetch(bp + 32, 0, 1);
      float4 w0 = *(const float4*)(bp);
      float4 w1 = *(const float4*)(bp + 4);
      float4 w2q = *(const float4*)(bp + 16);
      float4 w3 = *(const float4*)(bp + 20);

      float af[16] = {a0.x, a0.y, a0.z, a0.w, a1.x, a1.y, a1.z, a1.w,
                      a2.x, a2.y, a2.z, a2.w, a3.x, a3.y, a3.z, a3.w};
      float bfv[16] = {w0.x, w0.y, w0.z, w0.w, w1.x, w1.y, w1.z, w1.w,
                       w2q.x, w2q.y, w2q.z, w2q.w, w3.x, w3.y, w3.z, w3.w};
      v16bf ahi, alo, bhi, blo;
      #pragma unroll
      for (int i = 0; i < 16; ++i) {
        __bf16 h = (__bf16)af[i];
        ahi[i] = h;
        alo[i] = (__bf16)(af[i] - (float)h);
        __bf16 g = (__bf16)bfv[i];
        bhi[i] = g;
        blo[i] = (__bf16)(bfv[i] - (float)g);
      }
      acc = wmma_bf16(ahi, bhi, acc);
      acc = wmma_bf16(ahi, blo, acc);
      acc = wmma_bf16(alo, bhi, acc);
    }
    epilogue_tile(acc, b1[ex * kB + nn], W2[ex * kB + nn], rs);
  }

  if ((lane & 15) == 0) {
    const float b2v = b2[ex];
    #pragma unroll
    for (int r = 0; r < 8; ++r) {
      int m = mtile * 16 + half * 8 + r;
      if (m < cnt) out[bucket[ex * kN + m]] = rs[r] + b2v;
    }
  }
}

__global__ void finalize_kernel(const int* __restrict__ counts,
                                const float* __restrict__ l1part,
                                float* __restrict__ out_l1) {
  if (blockIdx.x == 0 && threadIdx.x == 0) {
    float tot = 0.0f;
    for (int e = 0; e < kE; ++e) {
      float s = 0.0f;
      for (int c = 0; c < 64; ++c) s += l1part[e * 64 + c];
      tot += (float)counts[e] * s;
    }
    *out_l1 = 1.0e-4f * tot;
  }
}

extern "C" void kernel_launch(void* const* d_in, const int* in_sizes, int n_in,
                              void* d_out, int out_size, void* d_ws, size_t ws_size,
                              hipStream_t stream) {
  const float* x  = (const float*)d_in[0];
  const int* keys = (const int*)d_in[1];
  const float* W1 = (const float*)d_in[2];
  const float* b1 = (const float*)d_in[3];
  const float* W2 = (const float*)d_in[4];
  const float* b2 = (const float*)d_in[5];
  float* out = (float*)d_out;

  // Workspace: [counts 16i][l1part 1024f][bucket 16*4096i] then (fast path)
  // bf16 hi/lo copies of x and W1 at a 512 KB base.
  int*   counts = (int*)d_ws;
  float* l1part = (float*)((char*)d_ws + 256);
  int*   bucket = (int*)((char*)d_ws + 256 + kE * 64 * sizeof(float));

  const size_t cvt_base = 512 * 1024;
  const size_t x_elems  = (size_t)kN * kD;
  const size_t w_elems  = (size_t)kE * kB * kD;
  __bf16* xh  = (__bf16*)((char*)d_ws + cvt_base);
  __bf16* xl  = xh + x_elems;
  __bf16* w1h = xl + x_elems;
  __bf16* w1l = w1h + w_elems;
  const size_t need = cvt_base + 2 * (x_elems + w_elems) * sizeof(__bf16);

  init_zero_kernel<<<(kN + 1 + 255) / 256, 256, 0, stream>>>(out, counts);
  scatter_kernel<<<kN / 256, 256, 0, stream>>>(keys, counts, bucket);
  l1_reduce_kernel<<<dim3(64, kE), 256, 0, stream>>>(W1, b1, W2, b2, l1part);

  if (ws_size >= need) {
    convert_kernel<<<(int)((x_elems / 4 + 255) / 256), 256, 0, stream>>>(x, xh, xl,
                                                                         (int)(x_elems / 4));
    convert_kernel<<<(int)((w_elems / 4 + 255) / 256), 256, 0, stream>>>(W1, w1h, w1l,
                                                                         (int)(w_elems / 4));
    moe_gemm_pre_kernel<<<dim3(kN / 16, kE), 32, 2 * 16 * kD * sizeof(__bf16), stream>>>(
        xh, xl, w1h, w1l, b1, W2, b2, counts, bucket, out);
  } else {
    moe_gemm_fly_kernel<<<dim3(kN / 16, kE), 32, 0, stream>>>(x, W1, b1, W2, b2,
                                                              counts, bucket, out);
  }
  finalize_kernel<<<1, 32, 0, stream>>>(counts, l1part, out + kN);
}
